// ScannedRNN_4088808866395
// MI455X (gfx1250) — matrix operand order repaired and checked by reading, not compiled, not run on hardware
//
#include <hip/hip_runtime.h>
#include <hip/hip_bf16.h>
#include <stdint.h>

#define TSTEPS 512
#define NB     128
#define HD     512
#define H3     1536
#define SCAN_BLOCKS 64   // 4 waves per block -> 256 scan units total

typedef __attribute__((ext_vector_type(16))) __bf16 v16bf;
typedef __attribute__((ext_vector_type(8)))  float  v8f;

static __device__ __forceinline__ unsigned short f2bf(float f) {
  unsigned u = __float_as_uint(f);
  unsigned r = u + 0x7FFFu + ((u >> 16) & 1u);   // round-to-nearest-even
  return (unsigned short)(r >> 16);
}

// A-operand (16x32 bf16): lane m (0-15) holds row m, elems 0..7 = K 0..7, 8..15 = K 16..23;
// lanes 16-31 hold K 8..15 / 24..31  (koff = 0 or 8)
static __device__ __forceinline__ v16bf load_a16(const unsigned short* p, int koff) {
  union { uint4 q[2]; v16bf v; } u;
  u.q[0] = *(const uint4*)(p + koff);
  u.q[1] = *(const uint4*)(p + koff + 16);
  return u.v;
}

// B-operand (32x16 bf16): lane n (0-15) holds col n, elems 0..7 = K 0..7, 8..15 = K 16..23; lanes 16-31 K 8..15 / 24..31
static __device__ __forceinline__ v16bf load_b16(const unsigned short* p, int koff16) {
  union { uint4 q[2]; v16bf v; } u;
  const unsigned short* q = p + (koff16 >> 1);
  u.q[0] = *(const uint4*)(q);
  u.q[1] = *(const uint4*)(q + 16);
  return u.v;
}

static __device__ __forceinline__ float sigmoidf_(float x) {
  return 1.0f / (1.0f + __expf(-x));
}

// ---------------- prep kernels ----------------

__global__ void gru_prep_h(const float* __restrict__ h0, float* __restrict__ hf0,
                           unsigned short* __restrict__ hb0, unsigned int* __restrict__ cnt) {
  int idx = blockIdx.x * blockDim.x + threadIdx.x;
  if (idx == 0) *cnt = 0u;                        // reset scan barrier every launch
  for (; idx < NB * HD; idx += gridDim.x * blockDim.x) {
    float v = h0[idx];
    hf0[idx] = v;
    hb0[idx] = f2bf(v);
  }
}

__global__ void gru_prep_x(const float* __restrict__ x, unsigned short* __restrict__ xb) {
  size_t idx = (size_t)blockIdx.x * blockDim.x + threadIdx.x;
  const size_t total = (size_t)TSTEPS * NB * HD;
  for (; idx < total; idx += (size_t)gridDim.x * blockDim.x)
    xb[idx] = f2bf(x[idx]);
}

// WT[n*HD + k] = bf16(W[k*H3 + n])  (N-major so B-tiles are contiguous in K)
__global__ void gru_prep_wT(const float* __restrict__ W, unsigned short* __restrict__ WT) {
  int idx = blockIdx.x * blockDim.x + threadIdx.x;
  const int total = H3 * HD;
  for (; idx < total; idx += gridDim.x * blockDim.x) {
    int n = idx / HD, k = idx - n * HD;
    WT[idx] = f2bf(W[(size_t)k * H3 + n]);
  }
}

// ---------------- phase 1: gi = x @ Wi + bi  (bf16 WMMA, f32 acc) ----------------
// wave = one 16(M) x 64(N) strip, K = 512. units = (65536/16) * (1536/64) = 98304 waves.

__global__ void __launch_bounds__(256) gru_gi_gemm(const unsigned short* __restrict__ xb,
                                                   const unsigned short* __restrict__ WiT,
                                                   const float* __restrict__ bi,
                                                   float* __restrict__ gi) {
  const int lane = threadIdx.x & 31;
  const int wave = threadIdx.x >> 5;
  const int unit = blockIdx.x * 8 + wave;
  const int mt = unit / 24;            // 0..4095   (16-row tile of 65536 rows)
  const int nq = unit % 24;            // 0..23     (64-col quad of 1536 cols)
  const int l15  = lane & 15;
  const int koffA = (lane >> 4) * 8;
  const int koffB = (lane >> 4) * 16;

  const unsigned short* arow = xb + (size_t)(mt * 16 + l15) * HD;

  float bias[4];
  v8f acc[4];
#pragma unroll
  for (int t = 0; t < 4; ++t) {
    bias[t] = bi[nq * 64 + t * 16 + l15];
    acc[t] = (v8f){0.f, 0.f, 0.f, 0.f, 0.f, 0.f, 0.f, 0.f};
  }

#pragma unroll 4
  for (int kc = 0; kc < 16; ++kc) {
    const int kb = kc * 32;
    v16bf A = load_a16(arow + kb, koffA);
#pragma unroll
    for (int t = 0; t < 4; ++t) {
      const unsigned short* bcol = WiT + (size_t)(nq * 64 + t * 16 + l15) * HD + kb;
      v16bf B = load_b16(bcol, koffB);
      acc[t] = __builtin_amdgcn_wmma_f32_16x16x32_bf16(false, A, false, B,
                                                       (short)0, acc[t], false, false);
    }
  }

  const int mofs = (lane >> 4) * 8;     // C/D layout: m = mofs+v, n = l15
#pragma unroll
  for (int t = 0; t < 4; ++t) {
    const int n = nq * 64 + t * 16 + l15;
#pragma unroll
    for (int v = 0; v < 8; ++v) {
      const int m = mt * 16 + mofs + v;
      gi[(size_t)m * H3 + n] = acc[t][v] + bias[t];
    }
  }
}

// ---------------- phase 2: persistent GRU scan ----------------
// 64 blocks x 128 thr = 256 waves, 1 wave/SIMD -> full 1024-VGPR budget per wave.
// Wh B-tiles (48 x v16bf = 384 VGPRs) are loaded ONCE and held in registers for
// all 512 steps; the per-step inner loop is just A-loads + 48 WMMA.

__global__ void __launch_bounds__(128) gru_scan(const float* __restrict__ gi,
                                                const unsigned short* __restrict__ WhT,
                                                const float* __restrict__ bhn,
                                                float* __restrict__ hf0, float* __restrict__ hf1,
                                                unsigned short* __restrict__ hb0,
                                                unsigned short* __restrict__ hb1,
                                                float* __restrict__ out,
                                                unsigned int* __restrict__ cnt) {
  const int lane  = threadIdx.x & 31;
  const int wave  = threadIdx.x >> 5;
  const int unit  = blockIdx.x * 4 + wave;   // 0..255
  const int mt    = unit >> 5;               // 0..7
  const int strip = unit & 31;               // 0..31
  const int l15   = lane & 15;
  const int koffA = (lane >> 4) * 8;
  const int koffB = (lane >> 4) * 16;
  const int ncol  = strip * 16 + l15;
  const int mofs  = (lane >> 4) * 8;

  const float bhn_l = bhn[ncol];
  float*          hf[2] = {hf0, hf1};
  unsigned short* hb[2] = {hb0, hb1};
  float* ys = out + (size_t)NB * HD;         // outputs after final-h slot

  // ---- preload all Wh tiles for this strip into registers (loop-invariant) ----
  v16bf Bw[48];
#pragma unroll
  for (int g = 0; g < 3; ++g) {
    const unsigned short* wp = WhT + (size_t)(g * HD + ncol) * HD;
#pragma unroll
    for (int kc = 0; kc < 16; ++kc)
      Bw[g * 16 + kc] = load_b16(wp + kc * 32, koffB);
  }

  int cur = 0;
#pragma unroll 1
  for (int t = 0; t < TSTEPS; ++t) {
    // prefetch next step's gi tile (streams from HBM) under this step's compute
    if (t + 1 < TSTEPS) {
      const float* gnext = gi + ((size_t)(t + 1) * NB + mt * 16 + mofs) * H3 + ncol;
#pragma unroll
      for (int v = 0; v < 8; ++v) {
        __builtin_prefetch(gnext + (size_t)v * H3, 0, 0);
        __builtin_prefetch(gnext + (size_t)v * H3 + HD, 0, 0);
        __builtin_prefetch(gnext + (size_t)v * H3 + 2 * HD, 0, 0);
      }
    }

    v8f ar = (v8f){0.f,0.f,0.f,0.f,0.f,0.f,0.f,0.f};
    v8f az = ar, an = ar;
    const unsigned short* hrow = hb[cur] + (size_t)(mt * 16 + l15) * HD;

#pragma unroll
    for (int kc = 0; kc < 16; ++kc) {
      v16bf A = load_a16(hrow + kc * 32, koffA);
      ar = __builtin_amdgcn_wmma_f32_16x16x32_bf16(false, A, false, Bw[kc],      (short)0, ar, false, false);
      az = __builtin_amdgcn_wmma_f32_16x16x32_bf16(false, A, false, Bw[16 + kc], (short)0, az, false, false);
      an = __builtin_amdgcn_wmma_f32_16x16x32_bf16(false, A, false, Bw[32 + kc], (short)0, an, false, false);
    }

    const int nxt = cur ^ 1;
#pragma unroll
    for (int v = 0; v < 8; ++v) {
      const int m = mt * 16 + mofs + v;
      const float* girow = gi + ((size_t)t * NB + m) * H3;
      const float r  = sigmoidf_(girow[ncol]          + ar[v]);
      const float z  = sigmoidf_(girow[HD + ncol]     + az[v]);
      const float n  = tanhf(girow[2 * HD + ncol] + r * (an[v] + bhn_l));
      const float hp = hf[cur][(size_t)m * HD + ncol];
      const float hn = (1.0f - z) * n + z * hp;
      hf[nxt][(size_t)m * HD + ncol] = hn;
      hb[nxt][(size_t)m * HD + ncol] = f2bf(hn);
      ys[(size_t)t * NB * HD + (size_t)m * HD + ncol] = hn;
    }

    // device-wide barrier: every wave flushes its stores device-scope first,
    // then one thread per block signals/spins on the monotonic counter.
    __threadfence();
    __syncthreads();
    if (threadIdx.x == 0) {
      __hip_atomic_fetch_add(cnt, 1u, __ATOMIC_RELEASE, __HIP_MEMORY_SCOPE_AGENT);
      const unsigned target = (unsigned)(SCAN_BLOCKS * (t + 1));
      while (__hip_atomic_load(cnt, __ATOMIC_ACQUIRE, __HIP_MEMORY_SCOPE_AGENT) < target)
        __builtin_amdgcn_s_sleep(1);
    }
    __syncthreads();
    __threadfence();
    cur ^= 1;
  }

  // final carry h_T -> out[0 .. B*H)
#pragma unroll
  for (int v = 0; v < 8; ++v) {
    const int m = mt * 16 + mofs + v;
    out[(size_t)m * HD + ncol] = hf[cur][(size_t)m * HD + ncol];
  }
}

// ---------------- launcher ----------------

extern "C" void kernel_launch(void* const* d_in, const int* in_sizes, int n_in,
                              void* d_out, int out_size, void* d_ws, size_t ws_size,
                              hipStream_t stream) {
  const float* x   = (const float*)d_in[0];
  const float* h0  = (const float*)d_in[1];
  const float* Wi  = (const float*)d_in[2];
  const float* bi  = (const float*)d_in[3];
  const float* Wh  = (const float*)d_in[4];
  const float* bhn = (const float*)d_in[5];
  float* out = (float*)d_out;

  char* w = (char*)d_ws;
  size_t o = 0;
  float* gi           = (float*)(w + o);          o += (size_t)TSTEPS * NB * H3 * sizeof(float); // 402.7 MB
  unsigned short* xb  = (unsigned short*)(w + o); o += (size_t)TSTEPS * NB * HD * 2;             // 67.1 MB
  unsigned short* WiT = (unsigned short*)(w + o); o += (size_t)H3 * HD * 2;
  unsigned short* WhT = (unsigned short*)(w + o); o += (size_t)H3 * HD * 2;
  float* hf0          = (float*)(w + o);          o += (size_t)NB * HD * 4;
  float* hf1          = (float*)(w + o);          o += (size_t)NB * HD * 4;
  unsigned short* hb0 = (unsigned short*)(w + o); o += (size_t)NB * HD * 2;
  unsigned short* hb1 = (unsigned short*)(w + o); o += (size_t)NB * HD * 2;
  unsigned int* cnt   = (unsigned int*)(w + o);

  gru_prep_h<<<256, 256, 0, stream>>>(h0, hf0, hb0, cnt);
  gru_prep_x<<<4096, 256, 0, stream>>>(x, xb);
  gru_prep_wT<<<1024, 256, 0, stream>>>(Wi, WiT);
  gru_prep_wT<<<1024, 256, 0, stream>>>(Wh, WhT);

  // 65536/16 M-tiles * 1536/64 N-quads = 98304 waves / 8 waves per block
  gru_gi_gemm<<<12288, 256, 0, stream>>>(xb, WiT, bi, gi);

  gru_scan<<<SCAN_BLOCKS, 128, 0, stream>>>(gi, WhT, bhn, hf0, hf1, hb0, hb1, out, cnt);
}
